// Dancer_48266842472922
// MI455X (gfx1250) — matrix operand order repaired and checked by reading, not compiled, not run on hardware
//
#include <hip/hip_runtime.h>
#include <stdint.h>

// ---------------- constants ----------------
#define NBLK  32
#define NTHR  256

#define B_    64
#define T_    240
#define DIM_  512
#define U_    1024
#define OUT_  75
#define DIN_K 640   // [curr(75) | _h(512) | zero-pad(53)] -> 20 K-chunks of 32

typedef __bf16 bf16_t;
typedef bf16_t v16bf __attribute__((ext_vector_type(16)));
typedef float  v8f   __attribute__((ext_vector_type(8)));

union Frag { v16bf v; uint4 q[2]; };

__device__ __forceinline__ v8f zero8() {
  v8f z;
#pragma unroll
  for (int i = 0; i < 8; ++i) z[i] = 0.0f;
  return z;
}

__device__ __forceinline__ bf16_t f2bf(float f) {
  uint32_t u = __builtin_bit_cast(uint32_t, f);
  u += 0x7FFFu + ((u >> 16) & 1u);            // round-to-nearest-even
  uint16_t h = (uint16_t)(u >> 16);
  return __builtin_bit_cast(bf16_t, h);
}

__device__ __forceinline__ float sigmoidf_(float x) {
  return 1.0f / (1.0f + __expf(-x));
}
__device__ __forceinline__ float tanhf_(float x) {
  x = fminf(fmaxf(x, -15.0f), 15.0f);
  float e = __expf(2.0f * x);
  return (e - 1.0f) / (e + 1.0f);
}

// ---------------- parameter block ----------------
struct CellP {
  const bf16_t* wxt;   // [4096, kxc*32]  transposed Wx, bf16
  const bf16_t* wht;   // [4096, 1024]    transposed Wh, bf16
  const float*  bias;  // [4096]
  int           kxc;   // #32-wide K chunks of x input
};

struct Params {
  CellP         cell[6];        // enc1,enc2,enc3,dec1,dec2,dec3
  const bf16_t* xa;             // [B, T*DIM] bf16 audio
  bf16_t*       h[6][2];        // ping-pong hidden, [B, U] bf16
  float*        c[6];           // cell state, [B, U] f32
  bf16_t*       din;            // [B, DIN_K] bf16 decoder input
  const bf16_t* fc_wt;          // [512, 1024]
  const float*  fc_b;
  const bf16_t* out_wt;         // [80, 1024] (rows 75..79 zero)
  const float*  out_b;
  float*        y;              // [B, T, 75] output
  int*          cnt;
  int*          gen;
};

// ---------------- grid barrier ----------------
__device__ __forceinline__ void grid_barrier(int* cnt, int* gen) {
  __threadfence();
  __syncthreads();
  if (threadIdx.x == 0) {
    int g = __hip_atomic_load(gen, __ATOMIC_RELAXED, __HIP_MEMORY_SCOPE_AGENT);
    int a = __hip_atomic_fetch_add(cnt, 1, __ATOMIC_ACQ_REL, __HIP_MEMORY_SCOPE_AGENT);
    if (a == NBLK - 1) {
      __hip_atomic_store(cnt, 0, __ATOMIC_RELAXED, __HIP_MEMORY_SCOPE_AGENT);
      __hip_atomic_fetch_add(gen, 1, __ATOMIC_RELEASE, __HIP_MEMORY_SCOPE_AGENT);
    } else {
      while (__hip_atomic_load(gen, __ATOMIC_ACQUIRE, __HIP_MEMORY_SCOPE_AGENT) == g) {
        __builtin_amdgcn_s_sleep(1);
      }
    }
  }
  __syncthreads();
  __threadfence();
}

// ---------------- WMMA K-loop over 4 gate tiles ----------------
// arow_base already includes row offset and laneHi*8.
__device__ __forceinline__ void kloop4(v8f acc[4],
    const bf16_t* __restrict__ arow_base, int kchunks,
    const bf16_t* __restrict__ wt, int ldw, int nrow0, int laneHi)
{
  for (int kc = 0; kc < kchunks; ++kc) {
    Frag a;
    const bf16_t* pa = arow_base + kc * 32;
    a.q[0] = *reinterpret_cast<const uint4*>(pa);        // K k0..k0+7   (lane half)
    a.q[1] = *reinterpret_cast<const uint4*>(pa + 16);   // K k0+16..+23 (lane half)
#pragma unroll
    for (int g = 0; g < 4; ++g) {
      Frag b;
      const bf16_t* pb = wt + (size_t)(g * U_ + nrow0) * ldw + kc * 32 + laneHi * 8;
      b.q[0] = *reinterpret_cast<const uint4*>(pb);
      b.q[1] = *reinterpret_cast<const uint4*>(pb + 16);
      acc[g] = __builtin_amdgcn_wmma_f32_16x16x32_bf16(
          false, a.v, false, b.v, (short)0, acc[g], false, false);
    }
  }
}

// ---------------- fused LSTM cell stage ----------------
__device__ void cell_stage(int wave, int laneLo, int laneHi,
    const bf16_t* __restrict__ x, int ldx, const CellP& cp,
    const bf16_t* __restrict__ hin, float* __restrict__ c, bf16_t* __restrict__ hout)
{
  const int mt = wave & 3;
  const int ng = wave >> 2;          // 0..63
  const int m0 = mt * 16;
  const int nrow0 = ng * 16 + laneLo;
  const int arow = m0 + laneLo;

  v8f acc[4];
#pragma unroll
  for (int g = 0; g < 4; ++g) acc[g] = zero8();

  const bf16_t* ax = x + (size_t)arow * ldx + laneHi * 8;
  kloop4(acc, ax, cp.kxc, cp.wxt, cp.kxc * 32, nrow0, laneHi);
  const bf16_t* ah = hin + (size_t)arow * U_ + laneHi * 8;
  kloop4(acc, ah, U_ / 32, cp.wht, U_, nrow0, laneHi);

  const float bi = cp.bias[nrow0];
  const float bf = cp.bias[U_ + nrow0];
  const float bg = cp.bias[2 * U_ + nrow0];
  const float bo = cp.bias[3 * U_ + nrow0];
#pragma unroll
  for (int r = 0; r < 8; ++r) {
    const int row = m0 + r + 8 * laneHi;
    const size_t off = (size_t)row * U_ + nrow0;
    float ig = sigmoidf_(acc[0][r] + bi);
    float fg = sigmoidf_(acc[1][r] + bf);
    float gg = tanhf_(acc[2][r] + bg);
    float og = sigmoidf_(acc[3][r] + bo);
    float cn = fg * c[off] + ig * gg;
    c[off] = cn;
    hout[off] = f2bf(og * tanhf_(cn));
  }
}

// ---------------- fc bottleneck: tanh(h @ fc_W + b) -> din[:,75:587] ----------------
__device__ void fc_stage(int wave, int laneLo, int laneHi,
    const bf16_t* __restrict__ in, const bf16_t* __restrict__ wt,
    const float* __restrict__ bias, bf16_t* __restrict__ din)
{
  if (wave >= 128) return;
  const int mt = wave & 3;
  const int ng = wave >> 2;          // 0..31
  const int m0 = mt * 16;
  const int nrow0 = ng * 16 + laneLo;
  const int arow = m0 + laneLo;

  v8f acc = zero8();
  const bf16_t* ax = in + (size_t)arow * U_ + laneHi * 8;
  for (int kc = 0; kc < U_ / 32; ++kc) {
    Frag a;
    const bf16_t* pa = ax + kc * 32;
    a.q[0] = *reinterpret_cast<const uint4*>(pa);
    a.q[1] = *reinterpret_cast<const uint4*>(pa + 16);
    Frag b;
    const bf16_t* pb = wt + (size_t)nrow0 * U_ + kc * 32 + laneHi * 8;
    b.q[0] = *reinterpret_cast<const uint4*>(pb);
    b.q[1] = *reinterpret_cast<const uint4*>(pb + 16);
    acc = __builtin_amdgcn_wmma_f32_16x16x32_bf16(
        false, a.v, false, b.v, (short)0, acc, false, false);
  }
  const float bb = bias[nrow0];
#pragma unroll
  for (int r = 0; r < 8; ++r) {
    const int row = m0 + r + 8 * laneHi;
    din[(size_t)row * DIN_K + OUT_ + nrow0] = f2bf(tanhf_(acc[r] + bb));
  }
}

// ---------------- output layer: elu(h @ out_W + b) -> y[:,t,:], din[:,0:75] ----------------
__device__ void out_stage(int wave, int laneLo, int laneHi, int t,
    const bf16_t* __restrict__ in, const bf16_t* __restrict__ wt,
    const float* __restrict__ bias, float* __restrict__ y, bf16_t* __restrict__ din)
{
  if (wave >= 20) return;
  const int mt = wave & 3;
  const int ng = wave >> 2;          // 0..4
  const int m0 = mt * 16;
  const int nrow0 = ng * 16 + laneLo; // 0..79
  const int arow = m0 + laneLo;

  v8f acc = zero8();
  const bf16_t* ax = in + (size_t)arow * U_ + laneHi * 8;
  for (int kc = 0; kc < U_ / 32; ++kc) {
    Frag a;
    const bf16_t* pa = ax + kc * 32;
    a.q[0] = *reinterpret_cast<const uint4*>(pa);
    a.q[1] = *reinterpret_cast<const uint4*>(pa + 16);
    Frag b;
    const bf16_t* pb = wt + (size_t)nrow0 * U_ + kc * 32 + laneHi * 8;
    b.q[0] = *reinterpret_cast<const uint4*>(pb);
    b.q[1] = *reinterpret_cast<const uint4*>(pb + 16);
    acc = __builtin_amdgcn_wmma_f32_16x16x32_bf16(
        false, a.v, false, b.v, (short)0, acc, false, false);
  }
  if (nrow0 < OUT_) {
    const float bb = bias[nrow0];
#pragma unroll
    for (int r = 0; r < 8; ++r) {
      const int row = m0 + r + 8 * laneHi;
      float v = acc[r] + bb;
      float e = v > 0.0f ? v : (__expf(v) - 1.0f);   // elu, alpha=1
      y[((size_t)row * T_ + t) * OUT_ + nrow0] = e;
      din[(size_t)row * DIN_K + nrow0] = f2bf(e);
    }
  }
}

// ---------------- persistent sequence kernel ----------------
__global__ __launch_bounds__(NTHR) void lstm_seq(Params p) {
  const int wave   = (blockIdx.x * NTHR + threadIdx.x) >> 5;
  const int lane   = threadIdx.x & 31;
  const int laneLo = lane & 15;
  const int laneHi = lane >> 4;

#pragma unroll 1
  for (int t = 0; t < T_; ++t) {
    const int rd = t & 1, wr = rd ^ 1;

    cell_stage(wave, laneLo, laneHi, p.xa + (size_t)t * DIM_, T_ * DIM_,
               p.cell[0], p.h[0][rd], p.c[0], p.h[0][wr]);
    grid_barrier(p.cnt, p.gen);

    cell_stage(wave, laneLo, laneHi, p.h[0][wr], U_,
               p.cell[1], p.h[1][rd], p.c[1], p.h[1][wr]);
    grid_barrier(p.cnt, p.gen);

    cell_stage(wave, laneLo, laneHi, p.h[1][wr], U_,
               p.cell[2], p.h[2][rd], p.c[2], p.h[2][wr]);
    grid_barrier(p.cnt, p.gen);

    fc_stage(wave, laneLo, laneHi, p.h[2][wr], p.fc_wt, p.fc_b, p.din);
    grid_barrier(p.cnt, p.gen);

    cell_stage(wave, laneLo, laneHi, p.din, DIN_K,
               p.cell[3], p.h[3][rd], p.c[3], p.h[3][wr]);
    grid_barrier(p.cnt, p.gen);

    cell_stage(wave, laneLo, laneHi, p.h[3][wr], U_,
               p.cell[4], p.h[4][rd], p.c[4], p.h[4][wr]);
    grid_barrier(p.cnt, p.gen);

    cell_stage(wave, laneLo, laneHi, p.h[4][wr], U_,
               p.cell[5], p.h[5][rd], p.c[5], p.h[5][wr]);
    grid_barrier(p.cnt, p.gen);

    out_stage(wave, laneLo, laneHi, t, p.h[5][wr], p.out_wt, p.out_b, p.y, p.din);
    grid_barrier(p.cnt, p.gen);
  }
}

// ---------------- prep kernels ----------------
// dst[n*Kout + k] = (k<K && n<N) ? bf16(src[k*N + n]) : 0
__global__ void k_transpose_cast(const float* __restrict__ src, bf16_t* __restrict__ dst,
                                 int K, int N, int Kout, int Nout) {
  size_t total = (size_t)Kout * Nout;
  for (size_t i = (size_t)blockIdx.x * blockDim.x + threadIdx.x; i < total;
       i += (size_t)gridDim.x * blockDim.x) {
    int k = (int)(i % (size_t)Kout);
    int n = (int)(i / (size_t)Kout);
    float v = (k < K && n < N) ? src[(size_t)k * N + n] : 0.0f;
    dst[i] = f2bf(v);
  }
}

__global__ void k_cast(const float* __restrict__ src, bf16_t* __restrict__ dst, size_t n) {
  for (size_t i = (size_t)blockIdx.x * blockDim.x + threadIdx.x; i < n;
       i += (size_t)gridDim.x * blockDim.x) {
    dst[i] = f2bf(src[i]);
  }
}

__global__ void k_init_din(const float* __restrict__ curr, bf16_t* __restrict__ din) {
  int i = blockIdx.x * blockDim.x + threadIdx.x;
  if (i < B_ * DIN_K) {
    int b = i / DIN_K, j = i % DIN_K;
    float v = (j < OUT_) ? curr[b * OUT_ + j] : 0.0f;
    din[i] = f2bf(v);
  }
}

// ---------------- host launcher ----------------
extern "C" void kernel_launch(void* const* d_in, const int* in_sizes, int n_in,
                              void* d_out, int out_size, void* d_ws, size_t ws_size,
                              hipStream_t stream) {
  (void)in_sizes; (void)n_in; (void)out_size; (void)ws_size;

  uintptr_t cur = ((uintptr_t)d_ws + 255) & ~(uintptr_t)255;
  auto alloc = [&](size_t bytes) -> void* {
    void* p = (void*)cur;
    cur = (cur + bytes + 255) & ~(uintptr_t)255;
    return p;
  };

  auto tco = [&](int idx, int K, int N, int Kout, int Nout) -> bf16_t* {
    bf16_t* d = (bf16_t*)alloc((size_t)Kout * Nout * sizeof(bf16_t));
    size_t total = (size_t)Kout * Nout;
    int blocks = (int)((total + 255) / 256);
    if (blocks > 32768) blocks = 32768;  // grid-stride covers the rest
    k_transpose_cast<<<blocks, 256, 0, stream>>>((const float*)d_in[idx], d, K, N, Kout, Nout);
    return d;
  };

  Params P;
  // dict order: 0 in_audio, 1 curr_step, then per cell Wx,Wh,b; fc between enc3/dec1
  P.cell[0] = { tco(2, 512, 4096, 512, 4096),  tco(3, 1024, 4096, 1024, 4096),
                (const float*)d_in[4], 16 };
  P.cell[1] = { tco(5, 1024, 4096, 1024, 4096), tco(6, 1024, 4096, 1024, 4096),
                (const float*)d_in[7], 32 };
  P.cell[2] = { tco(8, 1024, 4096, 1024, 4096), tco(9, 1024, 4096, 1024, 4096),
                (const float*)d_in[10], 32 };
  P.fc_wt = tco(11, 1024, 512, 1024, 512);
  P.fc_b  = (const float*)d_in[12];
  P.cell[3] = { tco(13, 587, 4096, 640, 4096),  tco(14, 1024, 4096, 1024, 4096),
                (const float*)d_in[15], 20 };
  P.cell[4] = { tco(16, 1024, 4096, 1024, 4096), tco(17, 1024, 4096, 1024, 4096),
                (const float*)d_in[18], 32 };
  P.cell[5] = { tco(19, 1024, 4096, 1024, 4096), tco(20, 1024, 4096, 1024, 4096),
                (const float*)d_in[21], 32 };
  P.out_wt = tco(22, 1024, 75, 1024, 80);
  P.out_b  = (const float*)d_in[23];

  // audio -> bf16
  {
    size_t n = (size_t)B_ * T_ * DIM_;
    bf16_t* xa = (bf16_t*)alloc(n * sizeof(bf16_t));
    int blocks = (int)((n + 255) / 256);
    if (blocks > 32768) blocks = 32768;
    k_cast<<<blocks, 256, 0, stream>>>((const float*)d_in[0], xa, n);
    P.xa = xa;
  }

  // state buffers
  for (int ci = 0; ci < 6; ++ci) {
    for (int j = 0; j < 2; ++j) {
      P.h[ci][j] = (bf16_t*)alloc((size_t)B_ * U_ * sizeof(bf16_t));
      (void)hipMemsetAsync(P.h[ci][j], 0, (size_t)B_ * U_ * sizeof(bf16_t), stream);
    }
    P.c[ci] = (float*)alloc((size_t)B_ * U_ * sizeof(float));
    (void)hipMemsetAsync(P.c[ci], 0, (size_t)B_ * U_ * sizeof(float), stream);
  }

  P.din = (bf16_t*)alloc((size_t)B_ * DIN_K * sizeof(bf16_t));
  k_init_din<<<(B_ * DIN_K + 255) / 256, 256, 0, stream>>>((const float*)d_in[1], P.din);

  int* sync = (int*)alloc(256);
  (void)hipMemsetAsync(sync, 0, 2 * sizeof(int), stream);
  P.cnt = sync;
  P.gen = sync + 1;
  P.y = (float*)d_out;

  lstm_seq<<<NBLK, NTHR, 0, stream>>>(P);
}
